// SimpleMoELayer_1717986918824
// MI455X (gfx1250) — matrix operand order repaired and checked by reading, not compiled, not run on hardware
//
#include <hip/hip_runtime.h>
#include <cstdint>
#include <cstddef>

// Problem constants (match reference)
#define H_DIM 1024
#define E_NUM 8
#define F_DIM 4096
#define NTOK  4096      // B*S
#define CAP   4096      // per-expert worst-case token capacity
// GEMM tiling
#define TM 128
#define TN 128
#define KK 64           // K-slab per LDS tile (2 WMMA k-steps of 32)
#define LSTR 72         // LDS row stride (bf16): 144B rows, 16B-aligned chunks

typedef __attribute__((ext_vector_type(16))) __bf16 v16bf;
typedef __attribute__((ext_vector_type(4)))  __bf16 v4bf;
typedef __attribute__((ext_vector_type(8)))  float  v8f;
typedef int v4i __attribute__((vector_size(16)));   // matches async-LDS builtin param type

struct B32x2 { uint4 lo, hi; };
struct V8P   { v4bf a, b; };

#if __has_builtin(__builtin_amdgcn_global_load_async_to_lds_b128)
#define HAS_ASYNC_LDS 1
#else
#define HAS_ASYNC_LDS 0
#endif

// 16-byte global -> LDS copy: async DMA when available, direct copy otherwise.
static __device__ __forceinline__ void async_cp16(const __bf16* g, __bf16* l) {
#if HAS_ASYNC_LDS
  __builtin_amdgcn_global_load_async_to_lds_b128(
      (__attribute__((address_space(1))) v4i*)g,
      (__attribute__((address_space(3))) v4i*)l, 0, 0);
#else
  *(uint4*)l = *(const uint4*)g;
#endif
}
static __device__ __forceinline__ void async_wait0() {
#if HAS_ASYNC_LDS
#if __has_builtin(__builtin_amdgcn_s_wait_asynccnt)
  __builtin_amdgcn_s_wait_asynccnt(0);
#else
  asm volatile("s_wait_asynccnt 0" ::: "memory");
#endif
#endif
}

// Compose a 16-element bf16 WMMA fragment from two 16-byte LDS chunks.
static __device__ __forceinline__ v16bf make_frag(const __bf16* p0, const __bf16* p1) {
  B32x2 t;
  t.lo = *(const uint4*)p0;
  t.hi = *(const uint4*)p1;
  return __builtin_bit_cast(v16bf, t);
}

static __device__ __forceinline__ v4bf pack4(float4 v) {
  v4bf r = { (__bf16)v.x, (__bf16)v.y, (__bf16)v.z, (__bf16)v.w };
  return r;
}
static __device__ __forceinline__ uint4 pack8(float4 a, float4 b) {
  V8P t; t.a = pack4(a); t.b = pack4(b);
  return __builtin_bit_cast(uint4, t);
}

// ---------------- kernel 0: zero output + per-expert counters ----------------
__global__ void moe_init_kernel(float* __restrict__ out, int n4, int* __restrict__ cnt) {
  int i = blockIdx.x * blockDim.x + threadIdx.x;
  if (i < n4) ((float4*)out)[i] = make_float4(0.f, 0.f, 0.f, 0.f);
  if (blockIdx.x == 0 && threadIdx.x < E_NUM) cnt[threadIdx.x] = 0;
}

// ---------------- kernel 0b: x f32 -> bf16 (read once per launch) ----------------
__global__ void moe_cvt_x_kernel(const float* __restrict__ x, __bf16* __restrict__ xb, int n8) {
  int i = blockIdx.x * blockDim.x + threadIdx.x;
  if (i < n8) {
    const float4* p = (const float4*)x + (size_t)i * 2;
    *(uint4*)(xb + (size_t)i * 8) = pack8(p[0], p[1]);
  }
}

// ---------------- kernel 0c: per-expert transpose+convert: src[R][C] f32 -> dst[C][R] bf16 ----
__global__ __launch_bounds__(256) void moe_transpose_kernel(
    const float* __restrict__ src, __bf16* __restrict__ dst, int R, int C) {
  __shared__ float tile[32][33];
  const int e = blockIdx.z;
  src += (size_t)e * R * C;
  dst += (size_t)e * R * C;
  const int c0 = blockIdx.x * 32, r0 = blockIdx.y * 32;
  const int tx = threadIdx.x & 31, ty = threadIdx.x >> 5;    // 32 x 8
#pragma unroll
  for (int i = 0; i < 4; ++i)
    tile[ty + i * 8][tx] = src[(size_t)(r0 + ty + i * 8) * C + c0 + tx];
  __syncthreads();
#pragma unroll
  for (int i = 0; i < 4; ++i)
    dst[(size_t)(c0 + ty + i * 8) * R + r0 + tx] = (__bf16)tile[tx][ty + i * 8];
}

// ---------------- kernel 1: router (one wave32 per token) ----------------
__global__ __launch_bounds__(256) void moe_router_kernel(
    const float* __restrict__ x, const float* __restrict__ Wr,
    int* __restrict__ cnt, int* __restrict__ tok, float* __restrict__ gate) {
  const int lane = threadIdx.x & 31;
  const int wave = threadIdx.x >> 5;
  const int t = blockIdx.x * 8 + wave;
  if (t >= NTOK) return;

  float acc[E_NUM];
#pragma unroll
  for (int e = 0; e < E_NUM; ++e) acc[e] = 0.f;

  const float* xr = x + (size_t)t * H_DIM;
  for (int h = lane * 4; h < H_DIM; h += 128) {
    float4 xv = *(const float4*)(xr + h);
#pragma unroll
    for (int e = 0; e < E_NUM; ++e) {
      float4 wv = *(const float4*)(Wr + e * H_DIM + h);
      acc[e] += xv.x * wv.x + xv.y * wv.y + xv.z * wv.z + xv.w * wv.w;
    }
  }
#pragma unroll
  for (int e = 0; e < E_NUM; ++e)
#pragma unroll
    for (int off = 16; off; off >>= 1)
      acc[e] += __shfl_xor(acc[e], off, 32);

  if (lane == 0) {
    int e0 = 0; float v0 = acc[0];
    for (int e = 1; e < E_NUM; ++e) if (acc[e] > v0) { v0 = acc[e]; e0 = e; }
    int e1 = (e0 == 0) ? 1 : 0; float v1 = acc[e1];
    for (int e = 0; e < E_NUM; ++e)
      if (e != e0 && acc[e] > v1) { v1 = acc[e]; e1 = e; }
    float p1 = __expf(v1 - v0);
    float inv = 1.f / (1.f + p1);
    float p0 = inv; p1 *= inv;
    int pos0 = atomicAdd(&cnt[e0], 1);
    tok[e0 * CAP + pos0] = t; gate[e0 * CAP + pos0] = p0;
    int pos1 = atomicAdd(&cnt[e1], 1);
    tok[e1 * CAP + pos1] = t; gate[e1 * CAP + pos1] = p1;
  }
}

// ---------------- kernel 2: exclusive prefix of expert counts ----------------
__global__ void moe_prefix_kernel(const int* __restrict__ cnt, int* __restrict__ base) {
  if (threadIdx.x == 0) {
    int s = 0;
    for (int e = 0; e < E_NUM; ++e) { base[e] = s; s += cnt[e]; }
  }
}

// ---------------- kernel 3: h = gelu(x_perm @ W1 + b1) -> bf16 ----------------
// A: xb [NTOK][H] bf16 (gathered rows); B: wt1 [E][F][H] bf16 (row n=f, k=h contiguous)
__global__ __launch_bounds__(256) void moe_ffn1_kernel(
    const __bf16* __restrict__ xb, const __bf16* __restrict__ wt1, const float* __restrict__ b1,
    const int* __restrict__ cnt, const int* __restrict__ base,
    const int* __restrict__ tok, __bf16* __restrict__ hbuf) {
  const int e    = blockIdx.x / (CAP / TM);
  const int mt   = blockIdx.x % (CAP / TM);
  const int c    = cnt[e];
  const int row0 = mt * TM;
  if (row0 >= c) return;                    // uniform: safe around WMMA
  const int f0 = blockIdx.y * TN;
  const int b0 = base[e];

  __shared__ __bf16 Xs[2][TM * LSTR];
  __shared__ __bf16 Ws[2][TN * LSTR];

  const int tid  = threadIdx.x;
  const int lane = tid & 31;
  const int wave = tid >> 5;
  const int wm   = (wave >> 1) * 32;        // 4 waves along M
  const int wn   = (wave & 1) * 64;         // 2 waves along N

  // --- hoisted gather descriptors (clamped => always valid) ---
  const int sm = tid >> 1;                  // staged row (2 threads per row)
  const int sc = (tid & 1) * 32;            // bf16 column base within KK
  const int xrow = (row0 + sm < c) ? (row0 + sm) : (c - 1);
  const __bf16* xptr = xb + (size_t)tok[e * CAP + xrow] * H_DIM + sc;
  const __bf16* wptr = wt1 + ((size_t)e * F_DIM + f0 + sm) * H_DIM + sc;
  __bf16* xls = &Xs[0][sm * LSTR + sc];
  __bf16* wls = &Ws[0][sm * LSTR + sc];
  const int bufoff = TM * LSTR;             // element offset between double buffers

  v8f acc[2][4] = {};
  const int akc = (lane >> 4) * 8;          // A: lanes<16 K{0..7,16..23}; >=16 K{8..15,24..31}
  const int bkc = (lane >> 4) * 8;          // B: lanes<16 K{0..7,16..23}; >=16 K{8..15,24..31}
  const int l15 = lane & 15;

  // prologue: async-stage tile 0 into buffer 0
#pragma unroll
  for (int i = 0; i < 4; ++i) async_cp16(xptr + i * 8, xls + i * 8);
#pragma unroll
  for (int i = 0; i < 4; ++i) async_cp16(wptr + i * 8, wls + i * 8);

  int buf = 0;
  for (int kt = 0; kt < H_DIM / KK; ++kt) {
    async_wait0();                          // tile kt resident in LDS[buf]
    __syncthreads();                        // publish; also retires readers of LDS[buf^1]

    if (kt + 1 < H_DIM / KK) {              // async-stage tile kt+1 into LDS[buf^1]
      const int h1 = (kt + 1) * KK;
      const int lo = (buf ^ 1) * bufoff;
#pragma unroll
      for (int i = 0; i < 4; ++i) async_cp16(xptr + h1 + i * 8, xls + lo + i * 8);
#pragma unroll
      for (int i = 0; i < 4; ++i) async_cp16(wptr + h1 + i * 8, wls + lo + i * 8);
      if (kt + 2 < H_DIM / KK)
        __builtin_prefetch(wptr + (size_t)(kt + 2) * KK, 0, 1);
    }

    // 32x64 per-wave register tile, two k=32 sub-steps: 16 WMMAs per barrier
#pragma unroll
    for (int ks = 0; ks < 2; ++ks) {
      const int kb = ks * 32;
#pragma unroll
      for (int i = 0; i < 2; ++i) {
        const __bf16* ap = &Xs[buf][(wm + i * 16 + l15) * LSTR + kb];
        v16bf a = make_frag(ap + akc, ap + akc + 16);
#pragma unroll
        for (int j = 0; j < 4; ++j) {
          const __bf16* bp = &Ws[buf][(wn + j * 16 + l15) * LSTR + kb];
          v16bf b = make_frag(bp + bkc, bp + bkc + 16);
          acc[i][j] = __builtin_amdgcn_wmma_f32_16x16x32_bf16(
              false, a, false, b, (short)0, acc[i][j], false, false);
        }
      }
    }
    buf ^= 1;
  }

  // epilogue: + b1, exact GeLU, store bf16
  const int hi8 = (lane >> 4) * 8;          // C/D layout: lanes>=16 hold M=r+8
#pragma unroll
  for (int i = 0; i < 2; ++i)
#pragma unroll
    for (int j = 0; j < 4; ++j) {
      int f = f0 + wn + j * 16 + (lane & 15);
      float bias = b1[e * F_DIM + f];
#pragma unroll
      for (int r = 0; r < 8; ++r) {
        int row = row0 + wm + i * 16 + r + hi8;
        if (row < c) {
          float v = acc[i][j][r] + bias;
          v = 0.5f * v * (1.f + erff(v * 0.70710678118654752f));
          hbuf[(size_t)(b0 + row) * F_DIM + f] = (__bf16)v;
        }
      }
    }
}

// ---------------- kernel 4: out += gate * (h @ W2 + b2) ----------------
// A: hbuf [rows][F] bf16; B: wt2 [E][H][F] bf16 (row n=h_out, k=f contiguous)
__global__ __launch_bounds__(256) void moe_ffn2_kernel(
    const __bf16* __restrict__ hbuf, const __bf16* __restrict__ wt2, const float* __restrict__ b2,
    const int* __restrict__ cnt, const int* __restrict__ base,
    const int* __restrict__ tok, const float* __restrict__ gate,
    float* __restrict__ out) {
  const int e    = blockIdx.x / (CAP / TM);
  const int mt   = blockIdx.x % (CAP / TM);
  const int c    = cnt[e];
  const int row0 = mt * TM;
  if (row0 >= c) return;
  const int h0 = blockIdx.y * TN;
  const int b0 = base[e];

  __shared__ __bf16 Hs[2][TM * LSTR];
  __shared__ __bf16 Ws[2][TN * LSTR];
  __shared__ int   ts[TM];
  __shared__ float gs[TM];

  const int tid  = threadIdx.x;
  const int lane = tid & 31;
  const int wave = tid >> 5;
  const int wm   = (wave >> 1) * 32;
  const int wn   = (wave & 1) * 64;

  // --- hoisted gather descriptors (clamped => always valid) ---
  const int sm = tid >> 1;
  const int sc = (tid & 1) * 32;
  const int hrow = (row0 + sm < c) ? (row0 + sm) : (c - 1);
  const __bf16* hptr = hbuf + (size_t)(b0 + hrow) * F_DIM + sc;
  const __bf16* wptr = wt2 + ((size_t)e * H_DIM + h0 + sm) * F_DIM + sc;
  __bf16* hls = &Hs[0][sm * LSTR + sc];
  __bf16* wls = &Ws[0][sm * LSTR + sc];
  const int bufoff = TM * LSTR;

  // stage tok/gate for this tile into LDS (epilogue use)
  if (tid < TM) {
    int row = row0 + tid;
    int rc = (row < c) ? row : (c - 1);
    ts[tid] = tok[e * CAP + rc];
    gs[tid] = gate[e * CAP + rc];
  }

  v8f acc[2][4] = {};
  const int akc = (lane >> 4) * 8;
  const int bkc = (lane >> 4) * 8;
  const int l15 = lane & 15;

  // prologue: async-stage tile 0 into buffer 0
#pragma unroll
  for (int i = 0; i < 4; ++i) async_cp16(hptr + i * 8, hls + i * 8);
#pragma unroll
  for (int i = 0; i < 4; ++i) async_cp16(wptr + i * 8, wls + i * 8);

  int buf = 0;
  for (int kt = 0; kt < F_DIM / KK; ++kt) {
    async_wait0();
    __syncthreads();

    if (kt + 1 < F_DIM / KK) {
      const int f1 = (kt + 1) * KK;
      const int lo = (buf ^ 1) * bufoff;
#pragma unroll
      for (int i = 0; i < 4; ++i) async_cp16(hptr + f1 + i * 8, hls + lo + i * 8);
#pragma unroll
      for (int i = 0; i < 4; ++i) async_cp16(wptr + f1 + i * 8, wls + lo + i * 8);
      if (kt + 2 < F_DIM / KK)
        __builtin_prefetch(wptr + (size_t)(kt + 2) * KK, 0, 1);
    }

#pragma unroll
    for (int ks = 0; ks < 2; ++ks) {
      const int kb = ks * 32;
#pragma unroll
      for (int i = 0; i < 2; ++i) {
        const __bf16* ap = &Hs[buf][(wm + i * 16 + l15) * LSTR + kb];
        v16bf a = make_frag(ap + akc, ap + akc + 16);
#pragma unroll
        for (int j = 0; j < 4; ++j) {
          const __bf16* bp = &Ws[buf][(wn + j * 16 + l15) * LSTR + kb];
          v16bf b = make_frag(bp + bkc, bp + bkc + 16);
          acc[i][j] = __builtin_amdgcn_wmma_f32_16x16x32_bf16(
              false, a, false, b, (short)0, acc[i][j], false, false);
        }
      }
    }
    buf ^= 1;
  }

  // epilogue: scale by gate, scatter-accumulate into out
  const int hi8 = (lane >> 4) * 8;
#pragma unroll
  for (int i = 0; i < 2; ++i)
#pragma unroll
    for (int j = 0; j < 4; ++j) {
      int hh = h0 + wn + j * 16 + (lane & 15);
      float bias = b2[e * H_DIM + hh];
#pragma unroll
      for (int r = 0; r < 8; ++r) {
        int m = wm + i * 16 + r + hi8;
        int row = row0 + m;
        if (row < c) {
          atomicAdd(&out[(size_t)ts[m] * H_DIM + hh], gs[m] * (acc[i][j][r] + bias));
        }
      }
    }
}

// ---------------- launch ----------------
extern "C" void kernel_launch(void* const* d_in, const int* in_sizes, int n_in,
                              void* d_out, int out_size, void* d_ws, size_t ws_size,
                              hipStream_t stream) {
  const float* x  = (const float*)d_in[0];
  const float* Wr = (const float*)d_in[1];
  const float* W1 = (const float*)d_in[2];
  const float* b1 = (const float*)d_in[3];
  const float* W2 = (const float*)d_in[4];
  const float* b2 = (const float*)d_in[5];
  float* out = (float*)d_out;

  // workspace layout (bytes)
  char* ws = (char*)d_ws;
  int*    cnt  = (int*)(ws + 0);                         // 8 ints
  int*    base = (int*)(ws + 256);                       // 8 ints
  int*    tok  = (int*)(ws + 1024);                      // E*CAP ints   (128 KB)
  float*  gate = (float*)(ws + 1024 + 4 * E_NUM * CAP);  // E*CAP floats (128 KB)
  __bf16* xb   = (__bf16*)(ws + (size_t)(1  << 20));     // 4096 x 1024 bf16   (8 MB)
  __bf16* hbuf = (__bf16*)(ws + (size_t)(16 << 20));     // 8192 x 4096 bf16  (64 MB)
  __bf16* wt1  = (__bf16*)(ws + (size_t)(80 << 20));     // E x [F][H] bf16   (64 MB)
  __bf16* wt2  = (__bf16*)(ws + (size_t)(144u << 20));   // E x [H][F] bf16   (64 MB)

  moe_init_kernel<<<(out_size / 4 + 255) / 256, 256, 0, stream>>>(out, out_size / 4, cnt);
  moe_cvt_x_kernel<<<(NTOK * H_DIM / 8 + 255) / 256, 256, 0, stream>>>(
      x, xb, NTOK * H_DIM / 8);
  {
    dim3 gt1(F_DIM / 32, H_DIM / 32, E_NUM);   // W1 [H][F] -> wt1 [F][H]
    moe_transpose_kernel<<<gt1, 256, 0, stream>>>(W1, wt1, H_DIM, F_DIM);
    dim3 gt2(H_DIM / 32, F_DIM / 32, E_NUM);   // W2 [F][H] -> wt2 [H][F]
    moe_transpose_kernel<<<gt2, 256, 0, stream>>>(W2, wt2, F_DIM, H_DIM);
  }
  moe_router_kernel<<<NTOK / 8, 256, 0, stream>>>(x, Wr, cnt, tok, gate);
  moe_prefix_kernel<<<1, 32, 0, stream>>>(cnt, base);
  dim3 g1(E_NUM * (CAP / TM), F_DIM / TN);
  moe_ffn1_kernel<<<g1, 256, 0, stream>>>(xb, wt1, b1, cnt, base, tok, hbuf);
  dim3 g2(E_NUM * (CAP / TM), H_DIM / TN);
  moe_ffn2_kernel<<<g2, 256, 0, stream>>>(hbuf, wt2, b2, cnt, base, tok, gate, out);
}
